// SelfAttention_43241730736912
// MI455X (gfx1250) — compile-verified, matrix operand order checked by reading, not run on hardware
//
#include <hip/hip_runtime.h>

typedef __attribute__((ext_vector_type(16))) __bf16 v16bf;
typedef __attribute__((ext_vector_type(8)))  __bf16 v8bf;
typedef __attribute__((ext_vector_type(8)))  float  v8f;

#define BB 4
#define TT 2048
#define CC 1024
#define HH 16
#define DHD 64

// Fragment loader for 16-bit WMMA operands (wave32).
// Lane L holds logical row/col (L&15); element j holds K = j + 8*((j>>3) + (L>>4)).
// Tiles are stored in LDS so that K is the contiguous (second) index -> two b128 loads.
__device__ __forceinline__ v16bf frag_ld(const __bf16* base, int stride, int row,
                                         int k0, int lane) {
  const int r  = row + (lane & 15);
  const int hi = lane >> 4;
  const __bf16* p = base + r * stride + k0 + 8 * hi;
  v8bf lo = *(const v8bf*)(p);        // K = 8*hi + 0..7
  v8bf hv = *(const v8bf*)(p + 16);   // K = 16 + 8*hi + 0..7
  v16bf o;
#pragma unroll
  for (int j = 0; j < 8; ++j) { o[j] = lo[j]; o[j + 8] = hv[j]; }
  return o;
}

__device__ __forceinline__ v8f wmma_bf16(v16bf a, v16bf b, v8f c) {
  return __builtin_amdgcn_wmma_f32_16x16x32_bf16(false, a, false, b, (short)0, c,
                                                 false, false);
}

// ---------------- Kernel 1: QKV projection -------------------------------
// y[b,h,t,d] = sum_c x[b,t,c] * W[h,c,d]  for W in {Wq,Wk,Wv}
// Block tile 128(M) x 64(N); 8 waves in 4x2 grid; each wave owns 2x2 16x16 tiles
// -> 4 WMMA per 4 fragment loads per 32-deep K step. Next K chunk is fetched
// into VGPRs while WMMAs run on the current LDS tile.
__global__ __launch_bounds__(256) void qkv_kernel(
    const float* __restrict__ x, const float* __restrict__ Wq,
    const float* __restrict__ Wk, const float* __restrict__ Wv,
    __bf16* __restrict__ qws, __bf16* __restrict__ kws, __bf16* __restrict__ vws) {
  __shared__ __bf16 As[128][40];   // [m][k]
  __shared__ __bf16 BsT[64][40];   // [n(d)][k(c)] transposed -> K contiguous
  const int tid = threadIdx.x, lane = tid & 31, w = tid >> 5;
  const int wr = w >> 1, wc = w & 1;
  const int bm0 = blockIdx.x * 128;
  const int sect = blockIdx.y >> 4;
  const int h = blockIdx.y & 15;
  const float* W = (sect == 0) ? Wq : (sect == 1) ? Wk : Wv;
  const float* Wh = W + (size_t)h * CC * DHD;
  __bf16* out = (sect == 0) ? qws : (sect == 1) ? kws : vws;

  const int arow = tid >> 1, ac0 = (tid & 1) * 16;  // A tile 128x32: 16 f32/thread
  const int wcc = tid >> 3, wd0 = (tid & 7) * 8;    // W tile 32x64 : 8 f32/thread

  float aR[16];
  float bR[8];
  auto fetch = [&](int kk) {
#pragma unroll
    for (int e = 0; e < 16; ++e)
      aR[e] = x[(size_t)(bm0 + arow) * CC + kk + ac0 + e];
#pragma unroll
    for (int e = 0; e < 8; ++e)
      bR[e] = Wh[(size_t)(kk + wcc) * DHD + wd0 + e];
  };
  fetch(0);

  v8f acc[2][2] = {};
  for (int kk = 0; kk < CC; kk += 32) {
    __syncthreads();  // previous compute done before overwriting LDS
    v8bf p0, p1;
#pragma unroll
    for (int e = 0; e < 8; ++e) { p0[e] = (__bf16)aR[e]; p1[e] = (__bf16)aR[8 + e]; }
    *(v8bf*)&As[arow][ac0] = p0;
    *(v8bf*)&As[arow][ac0 + 8] = p1;
#pragma unroll
    for (int e = 0; e < 8; ++e) BsT[wd0 + e][wcc] = (__bf16)bR[e];
    __syncthreads();
    if (kk + 32 < CC) fetch(kk + 32);  // hide global latency under WMMAs
    v16bf b0 = frag_ld(&BsT[0][0], 40, wc * 32, 0, lane);
    v16bf b1 = frag_ld(&BsT[0][0], 40, wc * 32 + 16, 0, lane);
    v16bf a0 = frag_ld(&As[0][0], 40, wr * 32, 0, lane);
    v16bf a1 = frag_ld(&As[0][0], 40, wr * 32 + 16, 0, lane);
    acc[0][0] = wmma_bf16(a0, b0, acc[0][0]);
    acc[0][1] = wmma_bf16(a0, b1, acc[0][1]);
    acc[1][0] = wmma_bf16(a1, b0, acc[1][0]);
    acc[1][1] = wmma_bf16(a1, b1, acc[1][1]);
  }

  // Scatter to [B,H,T,D]; the 128-row block never crosses a batch boundary,
  // so the base address is hoisted and per-element addressing is linear.
  const int hi = lane >> 4, nn = lane & 15;
  const int bq = bm0 >> 11, tb = bm0 & (TT - 1);
  const size_t obase = (((size_t)bq * HH + h) * TT + tb) * DHD;
#pragma unroll
  for (int rt = 0; rt < 2; ++rt)
#pragma unroll
    for (int ct = 0; ct < 2; ++ct) {
      const int d = wc * 32 + ct * 16 + nn;
#pragma unroll
      for (int r = 0; r < 8; ++r) {
        const int lr = wr * 32 + rt * 16 + 8 * hi + r;
        out[obase + (size_t)lr * DHD + d] = (__bf16)acc[rt][ct][r];
      }
    }
}

// ---------------- Kernel 2: flash attention ------------------------------
// grid = (T/64, B*H); block handles 64 queries, streams 64-key blocks.
// Q fragments are register-resident across the whole key loop.
__global__ __launch_bounds__(256) void attn_kernel(
    const __bf16* __restrict__ qws, const __bf16* __restrict__ kws,
    const __bf16* __restrict__ vws, __bf16* __restrict__ aws) {
  __shared__ __bf16 Qs[64][72];    // [q][d]
  __shared__ __bf16 Ks[64][72];    // [key][d]   (K contiguous for QK^T B-frag)
  __shared__ __bf16 VsT[64][72];   // [d][key]   (key contiguous for PV B-frag)
  __shared__ float  Sb[64][68];    // fp32 scores
  __shared__ __bf16 Pb[64][72];    // bf16 probabilities
  __shared__ float  mrow[64], lrow[64], arow[64];

  const int tid = threadIdx.x, lane = tid & 31, w = tid >> 5;
  const int qb = blockIdx.x, q0 = qb * 64;
  const int bh = blockIdx.y, b = bh >> 4, h = bh & 15;
  const size_t base = (size_t)bh * TT * DHD;
  const __bf16* Qg = qws + base;
  const __bf16* Kg = kws + base;
  const __bf16* Vg = vws + base;

  {
    int r = tid >> 2, d0 = (tid & 3) * 16;
    *(v8bf*)&Qs[r][d0]     = *(const v8bf*)&Qg[(size_t)(q0 + r) * DHD + d0];
    *(v8bf*)&Qs[r][d0 + 8] = *(const v8bf*)&Qg[(size_t)(q0 + r) * DHD + d0 + 8];
  }
  if (tid < 64) { mrow[tid] = -1e30f; lrow[tid] = 0.f; }

  const int trow = (w >> 2) * 16, tcol = (w & 3) * 16;
  const int hi = lane >> 4, nn = lane & 15;
  v8f o0 = {}, o1 = {};

  __syncthreads();

  v16bf qf[2][2];  // [kk/32][query-tile]; invariant over the key loop
#pragma unroll
  for (int kb = 0; kb < 2; ++kb) {
    qf[kb][0] = frag_ld(&Qs[0][0], 72, trow, kb * 32, lane);
    qf[kb][1] = frag_ld(&Qs[0][0], 72, trow + 32, kb * 32, lane);
  }

  for (int j = 0; j <= qb; ++j) {
    {  // stage K and V^T
      int r = tid >> 2, d0 = (tid & 3) * 16;
      *(v8bf*)&Ks[r][d0]     = *(const v8bf*)&Kg[(size_t)(j * 64 + r) * DHD + d0];
      *(v8bf*)&Ks[r][d0 + 8] = *(const v8bf*)&Kg[(size_t)(j * 64 + r) * DHD + d0 + 8];
#pragma unroll
      for (int e = 0; e < 16; ++e)
        VsT[d0 + e][r] = Vg[(size_t)(j * 64 + r) * DHD + d0 + e];
    }
    if (j < qb) {  // global_prefetch_b8 for the next key/value block
      __builtin_prefetch(&Kg[(size_t)(j + 1) * 64 * DHD], 0, 1);
      __builtin_prefetch(&Vg[(size_t)(j + 1) * 64 * DHD], 0, 1);
    }
    __syncthreads();

    // S = Q K^T (scaled by C^-0.5 = 1/32), causal mask
    v8f s0 = {}, s1 = {};
#pragma unroll
    for (int kb = 0; kb < 2; ++kb) {
      v16bf bf = frag_ld(&Ks[0][0], 72, tcol, kb * 32, lane);
      s0 = wmma_bf16(qf[kb][0], bf, s0);
      s1 = wmma_bf16(qf[kb][1], bf, s1);
    }
    const int key = j * 64 + tcol + nn;
#pragma unroll
    for (int r = 0; r < 8; ++r) {
      int ql0 = trow + 8 * hi + r;
      Sb[ql0][tcol + nn] = (key <= q0 + ql0) ? s0[r] * 0.03125f : -1e30f;
      int ql1 = ql0 + 32;
      Sb[ql1][tcol + nn] = (key <= q0 + ql1) ? s1[r] * 0.03125f : -1e30f;
    }
    __syncthreads();

    // online softmax (one row per thread for tid<64); Pb written 8-wide packed
    if (tid < 64) {
      float mold = mrow[tid], mnew = mold;
#pragma unroll 8
      for (int c = 0; c < 64; ++c) mnew = fmaxf(mnew, Sb[tid][c]);
      float al = __expf(mold - mnew);
      float s = 0.f;
#pragma unroll
      for (int c0 = 0; c0 < 64; c0 += 8) {
        v8bf pk;
#pragma unroll
        for (int e = 0; e < 8; ++e) {
          float p = __expf(Sb[tid][c0 + e] - mnew);
          pk[e] = (__bf16)p;
          s += p;
        }
        *(v8bf*)&Pb[tid][c0] = pk;
      }
      lrow[tid] = lrow[tid] * al + s;
      mrow[tid] = mnew;
      arow[tid] = al;
    }
    __syncthreads();

    // O = O*alpha + P V
#pragma unroll
    for (int r = 0; r < 8; ++r) {
      o0[r] *= arow[trow + 8 * hi + r];
      o1[r] *= arow[trow + 32 + 8 * hi + r];
    }
#pragma unroll
    for (int kb = 0; kb < 2; ++kb) {
      v16bf bf = frag_ld(&VsT[0][0], 72, tcol, kb * 32, lane);
      v16bf a0 = frag_ld(&Pb[0][0], 72, trow, kb * 32, lane);
      v16bf a1 = frag_ld(&Pb[0][0], 72, trow + 32, kb * 32, lane);
      o0 = wmma_bf16(a0, bf, o0);
      o1 = wmma_bf16(a1, bf, o1);
    }
    __syncthreads();
  }

  // write concatenated-head output [B,T,H*DH] (bf16); linear addressing
  const size_t abase = ((size_t)b * TT + q0) * CC + h * DHD + tcol + nn;
#pragma unroll
  for (int r = 0; r < 8; ++r) {
    const int ql0 = trow + 8 * hi + r;
    aws[abase + (size_t)ql0 * CC] = (__bf16)(o0[r] / lrow[ql0]);
    const int ql1 = ql0 + 32;
    aws[abase + (size_t)ql1 * CC] = (__bf16)(o1[r] / lrow[ql1]);
  }
}

// ---------------- Kernel 3: output projection ----------------------------
// Same 128x64 / 2x2-tiles-per-wave structure as kernel 1; A is already bf16.
__global__ __launch_bounds__(256) void proj_kernel(
    const __bf16* __restrict__ aws, const float* __restrict__ Wp,
    float* __restrict__ out) {
  __shared__ __bf16 As[128][40];
  __shared__ __bf16 BsT[64][40];
  const int tid = threadIdx.x, lane = tid & 31, w = tid >> 5;
  const int wr = w >> 1, wc = w & 1;
  const int bm0 = blockIdx.x * 128;
  const int n0 = blockIdx.y * 64;

  const int arow = tid >> 1, ac0 = (tid & 1) * 16;
  const int wcc = tid >> 3, wd0 = (tid & 7) * 8;

  v8bf aR0, aR1;
  float bR[8];
  auto fetch = [&](int kk) {
    aR0 = *(const v8bf*)&aws[(size_t)(bm0 + arow) * CC + kk + ac0];
    aR1 = *(const v8bf*)&aws[(size_t)(bm0 + arow) * CC + kk + ac0 + 8];
#pragma unroll
    for (int e = 0; e < 8; ++e)
      bR[e] = Wp[(size_t)(kk + wcc) * CC + n0 + wd0 + e];
  };
  fetch(0);

  v8f acc[2][2] = {};
  for (int kk = 0; kk < CC; kk += 32) {
    __syncthreads();
    *(v8bf*)&As[arow][ac0] = aR0;
    *(v8bf*)&As[arow][ac0 + 8] = aR1;
#pragma unroll
    for (int e = 0; e < 8; ++e) BsT[wd0 + e][wcc] = (__bf16)bR[e];
    __syncthreads();
    if (kk + 32 < CC) fetch(kk + 32);
    v16bf b0 = frag_ld(&BsT[0][0], 40, wc * 32, 0, lane);
    v16bf b1 = frag_ld(&BsT[0][0], 40, wc * 32 + 16, 0, lane);
    v16bf a0 = frag_ld(&As[0][0], 40, wr * 32, 0, lane);
    v16bf a1 = frag_ld(&As[0][0], 40, wr * 32 + 16, 0, lane);
    acc[0][0] = wmma_bf16(a0, b0, acc[0][0]);
    acc[0][1] = wmma_bf16(a0, b1, acc[0][1]);
    acc[1][0] = wmma_bf16(a1, b0, acc[1][0]);
    acc[1][1] = wmma_bf16(a1, b1, acc[1][1]);
  }

  const int hi = lane >> 4, nn = lane & 15;
  const size_t obase = (size_t)bm0 * CC + n0;
#pragma unroll
  for (int rt = 0; rt < 2; ++rt)
#pragma unroll
    for (int ct = 0; ct < 2; ++ct) {
      const int d = wc * 32 + ct * 16 + nn;
#pragma unroll
      for (int r = 0; r < 8; ++r) {
        const int lr = wr * 32 + rt * 16 + 8 * hi + r;
        out[obase + (size_t)lr * CC + d] = acc[rt][ct][r];
      }
    }
}

extern "C" void kernel_launch(void* const* d_in, const int* in_sizes, int n_in,
                              void* d_out, int out_size, void* d_ws, size_t ws_size,
                              hipStream_t stream) {
  const float* x  = (const float*)d_in[0];
  const float* Wq = (const float*)d_in[1];
  const float* Wk = (const float*)d_in[2];
  const float* Wv = (const float*)d_in[3];
  const float* Wp = (const float*)d_in[4];
  float* out = (float*)d_out;

  const size_t nq = (size_t)BB * HH * TT * DHD;  // 8.39M elements per tensor
  char* wsb = (char*)d_ws;
  __bf16* qws = (__bf16*)(wsb);
  __bf16* kws = (__bf16*)(wsb + 2 * nq);
  __bf16* vws = (__bf16*)(wsb + 4 * nq);
  __bf16* aws = (__bf16*)(wsb + 6 * nq);  // [B*T, C] bf16; total 64 MB

  dim3 blk(256);
  qkv_kernel<<<dim3(BB * TT / 128, 3 * HH), blk, 0, stream>>>(x, Wq, Wk, Wv, qws,
                                                              kws, vws);
  attn_kernel<<<dim3(TT / 64, BB * HH), blk, 0, stream>>>(qws, kws, vws, aws);
  proj_kernel<<<dim3(BB * TT / 128, CC / 64), blk, 0, stream>>>(aws, Wp, out);
}
